// AdaptiveSpanAttention_13219909337509
// MI455X (gfx1250) — matrix operand order repaired and checked by reading, not compiled, not run on hardware
//
#include <hip/hip_runtime.h>
#include <hip/hip_bf16.h>
#include <math.h>

#define BB 4
#define TT 2048
#define DD 512
#define HH 8
#define DH 64

typedef __bf16 bf16;
typedef __attribute__((ext_vector_type(16))) __bf16 v16bf;
typedef __attribute__((ext_vector_type(8)))  __bf16 v8bf;
typedef __attribute__((ext_vector_type(8)))  float  v8f;

__device__ __forceinline__ bf16 to_bf16(float f) {
    union { float f; unsigned u; } x; x.f = f;
    unsigned r = x.u + 0x7FFFu + ((x.u >> 16) & 1u);   // round-to-nearest-even
    union { unsigned short s; bf16 b; } y; y.s = (unsigned short)(r >> 16);
    return y.b;
}

__device__ __forceinline__ v16bf cat8(v8bf lo, v8bf hi) {
    return __builtin_shufflevector(lo, hi, 0,1,2,3,4,5,6,7,8,9,10,11,12,13,14,15);
}

// ---------------------------------------------------------------------------
// Kernel 0: one-shot fp32 -> bf16 conversion (x and the four weights).
// ---------------------------------------------------------------------------
__global__ __launch_bounds__(256) void cvt_bf16_kernel(
    const float* __restrict__ src, bf16* __restrict__ dst, int n)
{
    int i = (blockIdx.x * 256 + threadIdx.x) * 8;
    if (i >= n) return;
    v8f f = *(const v8f*)(src + i);
    v8bf o;
    #pragma unroll
    for (int j = 0; j < 8; ++j) o[j] = to_bf16(f[j]);
    *(v8bf*)(dst + i) = o;
}

// ---------------------------------------------------------------------------
// Kernel 1: QKV projection, Y = x @ W^T.  32x32 C block per wave:
// 2 A tiles x 2 B tiles -> 4 WMMA per K-step, 1 load per WMMA.
// Q gets the 1/sqrt(dh) scale folded in.  V is written transposed (b,h,dh,t).
// ---------------------------------------------------------------------------
__global__ __launch_bounds__(256) void qkv_proj_kernel(
    const bf16* __restrict__ xbf, const bf16* __restrict__ wqb,
    const bf16* __restrict__ wkb, const bf16* __restrict__ wvb,
    bf16* __restrict__ Qbf, bf16* __restrict__ Kbf, bf16* __restrict__ Vt)
{
    int wid  = blockIdx.x * 8 + (threadIdx.x >> 5);
    int lane = threadIdx.x & 31;
    if (wid >= 3 * 256 * 16) return;               // wave-uniform
    int mat = wid / (256 * 16);
    int rem = wid % (256 * 16);
    int m0 = (rem / 16) * 32;
    int n0 = (rem % 16) * 32;
    const bf16* W = (mat == 0) ? wqb : (mat == 1) ? wkb : wvb;
    int half = lane >> 4;
    int lidx = lane & 15;

    const bf16* arow0 = xbf + (size_t)(m0 + lidx) * DD;
    const bf16* arow1 = arow0 + (size_t)16 * DD;
    const bf16* brow0 = W   + (size_t)(n0 + lidx) * DD;
    const bf16* brow1 = brow0 + (size_t)16 * DD;

    v8f c[2][2] = {};
    for (int kb = 0; kb < DD; kb += 32) {
        v16bf a0 = *(const v16bf*)(arow0 + kb + half * 16);
        v16bf a1 = *(const v16bf*)(arow1 + kb + half * 16);
        v16bf b0 = *(const v16bf*)(brow0 + kb + half * 16);
        v16bf b1 = *(const v16bf*)(brow1 + kb + half * 16);
        c[0][0] = __builtin_amdgcn_wmma_f32_16x16x32_bf16(false, a0, false, b0, (short)0, c[0][0], false, false);
        c[0][1] = __builtin_amdgcn_wmma_f32_16x16x32_bf16(false, a0, false, b1, (short)0, c[0][1], false, false);
        c[1][0] = __builtin_amdgcn_wmma_f32_16x16x32_bf16(false, a1, false, b0, (short)0, c[1][0], false, false);
        c[1][1] = __builtin_amdgcn_wmma_f32_16x16x32_bf16(false, a1, false, b1, (short)0, c[1][1], false, false);
    }

    // uniform select of destination + strides: no divergent store path
    bf16* dst  = (mat == 0) ? Qbf : (mat == 1) ? Kbf : Vt;
    int   ts   = (mat == 2) ? 1  : DH;   // stride of t
    int   dstd = (mat == 2) ? TT : 1;    // stride of d
    float smul = (mat == 0) ? 0.125f : 1.0f;

    #pragma unroll
    for (int mi = 0; mi < 2; ++mi)
        #pragma unroll
        for (int ni = 0; ni < 2; ++ni) {
            int n = n0 + ni * 16 + lidx;
            int h = n >> 6, d = n & 63;
            #pragma unroll
            for (int r = 0; r < 8; ++r) {
                int m  = m0 + mi * 16 + r + 8 * half;
                int bb = m >> 11, t = m & 2047;
                dst[(size_t)(bb * HH + h) * (TT * DH) + (size_t)t * ts + (size_t)d * dstd] =
                    to_bf16(c[mi][ni][r] * smul);
            }
        }
}

// ---------------------------------------------------------------------------
// Kernel 2: adaptive-span flash attention. One wave per (b,h,16-query tile).
// Interior tiles (fully causal, fully inside the span plateau) skip all mask
// math; boundary/diagonal tiles take the log-ramp path.
// ---------------------------------------------------------------------------
__global__ __launch_bounds__(32) void attn_kernel(
    const bf16* __restrict__ Qbf, const bf16* __restrict__ Kbf,
    const bf16* __restrict__ Vt,  const float* __restrict__ span,
    bf16* __restrict__ Abf)
{
    __shared__ float p_lds[16][33];
    int lane = threadIdx.x & 31;
    int half = lane >> 4, lidx = lane & 15;
    int id = blockIdx.x;
    int qt = id & 127;
    int bh = id >> 7;
    int bb = bh >> 3, h = bh & 7;
    int q0 = qt * 16;
    float eff = fminf(fmaxf(span[h], 0.0f), 1.0f) * 512.0f;

    // Q tile in A-layout (two 16x32 K-slabs of dh); scale already folded in
    const bf16* qrow = Qbf + ((size_t)(bb * HH + h) * TT + (q0 + lidx)) * DH;
    v16bf aq0 = *(const v16bf*)(qrow +      half * 16);
    v16bf aq1 = *(const v16bf*)(qrow + 32 + half * 16);

    float m_run[8], l_run[8];
    v8f o[4] = {};
    #pragma unroll
    for (int r = 0; r < 8; ++r) { m_run[r] = -1e30f; l_run[r] = 0.0f; }

    int jmin = (int)floorf((float)q0 - eff - 1.0f);
    if (jmin < 0) jmin = 0;
    int kt0 = jmin >> 5;
    int kt1 = (q0 + 15) >> 5;

    const bf16* kmat = Kbf + (size_t)(bb * HH + h) * TT * DH;
    const bf16* vmat = Vt  + (size_t)(bb * HH + h) * DH * TT;

    for (int kt = kt0; kt <= kt1; ++kt) {
        int k0 = kt << 5;

        if (kt < kt1) {  // prefetch next K/V tiles (global_prefetch_b8)
            __builtin_prefetch(kmat + (size_t)(k0 + 32 + lane) * DH, 0, 0);
            __builtin_prefetch(vmat + (size_t)lane * TT + k0 + 32, 0, 0);
            __builtin_prefetch(vmat + (size_t)(32 + lane) * TT + k0 + 32, 0, 0);
        }

        // S = Q K^T for two 16-key column tiles
        v8f s[2];
        #pragma unroll
        for (int c2 = 0; c2 < 2; ++c2) {
            const bf16* krow = kmat + (size_t)(k0 + c2 * 16 + lidx) * DH;
            v16bf bkA = *(const v16bf*)(krow +      half * 16);
            v16bf bkB = *(const v16bf*)(krow + 32 + half * 16);
            v8f sc = {};
            sc = __builtin_amdgcn_wmma_f32_16x16x32_bf16(false, aq0, false, bkA, (short)0, sc, false, false);
            sc = __builtin_amdgcn_wmma_f32_16x16x32_bf16(false, aq1, false, bkB, (short)0, sc, false, false);
            s[c2] = sc;
        }

        // mask: skip entirely for interior tiles (wave-uniform test)
        float sv[2][8];
        bool interior = (k0 + 31 <= q0) && ((float)(q0 + 15 - k0) <= eff - 3.0f);
        if (interior) {
            #pragma unroll
            for (int c2 = 0; c2 < 2; ++c2)
                #pragma unroll
                for (int r = 0; r < 8; ++r) sv[c2][r] = s[c2][r];
        } else {
            #pragma unroll
            for (int c2 = 0; c2 < 2; ++c2)
                #pragma unroll
                for (int r = 0; r < 8; ++r) {
                    int i = q0 + r + 8 * half;
                    int j = k0 + c2 * 16 + lidx;
                    float dist = (float)(i - j);
                    float R = fminf(fmaxf((eff - dist + 1.0f) * 0.25f, 0.0f), 1.0f);
                    float mval = (i >= j && R > 0.0f) ? fmaxf(__logf(R + 1e-10f), -1.0e9f)
                                                      : -1.0e9f;
                    sv[c2][r] = s[c2][r] + mval;
                }
        }

        // online softmax (row = 16 lanes within a half)
        #pragma unroll
        for (int r = 0; r < 8; ++r) {
            float mx = fmaxf(sv[0][r], sv[1][r]);
            mx = fmaxf(mx, __shfl_xor(mx, 1, 32));
            mx = fmaxf(mx, __shfl_xor(mx, 2, 32));
            mx = fmaxf(mx, __shfl_xor(mx, 4, 32));
            mx = fmaxf(mx, __shfl_xor(mx, 8, 32));
            float nm   = fmaxf(m_run[r], mx);
            float corr = __expf(m_run[r] - nm);
            m_run[r] = nm;
            float p0 = __expf(sv[0][r] - nm);
            float p1 = __expf(sv[1][r] - nm);
            float rs = p0 + p1;
            rs += __shfl_xor(rs, 1, 32);
            rs += __shfl_xor(rs, 2, 32);
            rs += __shfl_xor(rs, 4, 32);
            rs += __shfl_xor(rs, 8, 32);
            l_run[r] = l_run[r] * corr + rs;
            #pragma unroll
            for (int c = 0; c < 4; ++c) o[c][r] *= corr;
            p_lds[r + 8 * half][lidx]      = p0;
            p_lds[r + 8 * half][16 + lidx] = p1;
        }
        __syncthreads();

        // P (16x32) C-layout -> A-layout bf16
        v16bf ap;
        #pragma unroll
        for (int v = 0; v < 8; ++v) {
            int kk = 2 * v + half * 16;
            ap[2 * v]     = to_bf16(p_lds[lidx][kk]);
            ap[2 * v + 1] = to_bf16(p_lds[lidx][kk + 1]);
        }
        __syncthreads();

        // O += P V : V^T stored (dh, t) so B-operand is contiguous along keys
        #pragma unroll
        for (int c = 0; c < 4; ++c) {
            const bf16* vrow = vmat + (size_t)(c * 16 + lidx) * TT + k0;
            v16bf bv = *(const v16bf*)(vrow + half * 16);
            o[c] = __builtin_amdgcn_wmma_f32_16x16x32_bf16(false, ap, false, bv, (short)0, o[c], false, false);
        }
    }

    // normalize and store attention output as (b, t, h*64+d) bf16
    #pragma unroll
    for (int r = 0; r < 8; ++r) {
        float inv = 1.0f / l_run[r];
        int i = q0 + r + 8 * half;
        size_t rowbase = (size_t)(bb * TT + i) * DD + h * DH;
        #pragma unroll
        for (int c = 0; c < 4; ++c)
            Abf[rowbase + c * 16 + lidx] = to_bf16(o[c][r] * inv);
    }
}

// ---------------------------------------------------------------------------
// Kernel 3: output projection, out = A @ wo^T (fp32 out), 32x32 per wave.
// ---------------------------------------------------------------------------
__global__ __launch_bounds__(256) void out_proj_kernel(
    const bf16* __restrict__ Abf, const bf16* __restrict__ wob,
    float* __restrict__ out)
{
    int wid  = blockIdx.x * 8 + (threadIdx.x >> 5);
    int lane = threadIdx.x & 31;
    if (wid >= 256 * 16) return;
    int m0 = (wid / 16) * 32;
    int n0 = (wid % 16) * 32;
    int half = lane >> 4, lidx = lane & 15;

    const bf16* arow0 = Abf + (size_t)(m0 + lidx) * DD;
    const bf16* arow1 = arow0 + (size_t)16 * DD;
    const bf16* brow0 = wob + (size_t)(n0 + lidx) * DD;
    const bf16* brow1 = brow0 + (size_t)16 * DD;

    v8f c[2][2] = {};
    for (int kb = 0; kb < DD; kb += 32) {
        v16bf a0 = *(const v16bf*)(arow0 + kb + half * 16);
        v16bf a1 = *(const v16bf*)(arow1 + kb + half * 16);
        v16bf b0 = *(const v16bf*)(brow0 + kb + half * 16);
        v16bf b1 = *(const v16bf*)(brow1 + kb + half * 16);
        c[0][0] = __builtin_amdgcn_wmma_f32_16x16x32_bf16(false, a0, false, b0, (short)0, c[0][0], false, false);
        c[0][1] = __builtin_amdgcn_wmma_f32_16x16x32_bf16(false, a0, false, b1, (short)0, c[0][1], false, false);
        c[1][0] = __builtin_amdgcn_wmma_f32_16x16x32_bf16(false, a1, false, b0, (short)0, c[1][0], false, false);
        c[1][1] = __builtin_amdgcn_wmma_f32_16x16x32_bf16(false, a1, false, b1, (short)0, c[1][1], false, false);
    }

    #pragma unroll
    for (int mi = 0; mi < 2; ++mi)
        #pragma unroll
        for (int ni = 0; ni < 2; ++ni) {
            int n = n0 + ni * 16 + lidx;
            #pragma unroll
            for (int r = 0; r < 8; ++r)
                out[(size_t)(m0 + mi * 16 + r + 8 * half) * DD + n] = c[mi][ni][r];
        }
}

__global__ void span_loss_kernel(const float* __restrict__ span, float* __restrict__ out) {
    if (threadIdx.x == 0 && blockIdx.x == 0) {
        float s = 0.0f;
        for (int i = 0; i < HH; ++i) s += fminf(fmaxf(span[i], 0.0f), 1.0f);
        out[0] = 2e-4f * (s / (float)HH);
    }
}

extern "C" void kernel_launch(void* const* d_in, const int* in_sizes, int n_in,
                              void* d_out, int out_size, void* d_ws, size_t ws_size,
                              hipStream_t stream) {
    const float* x    = (const float*)d_in[0];
    const float* wq   = (const float*)d_in[1];
    const float* wk   = (const float*)d_in[2];
    const float* wv   = (const float*)d_in[3];
    const float* wo   = (const float*)d_in[4];
    const float* span = (const float*)d_in[5];
    float* out = (float*)d_out;

    char* ws = (char*)d_ws;
    size_t segQ = (size_t)BB * HH * TT * DH * sizeof(bf16);   // 8 MB
    size_t segW = (size_t)DD * DD * sizeof(bf16);             // 512 KB
    bf16* Qbf = (bf16*)(ws);
    bf16* Kbf = (bf16*)(ws + segQ);
    bf16* Vt  = (bf16*)(ws + 2 * segQ);
    bf16* Abf = (bf16*)(ws + 3 * segQ);
    bf16* xbf = (bf16*)(ws + 4 * segQ);                       // 8 MB
    bf16* wqb = (bf16*)(ws + 5 * segQ);
    bf16* wkb = (bf16*)(ws + 5 * segQ + segW);
    bf16* wvb = (bf16*)(ws + 5 * segQ + 2 * segW);
    bf16* wob = (bf16*)(ws + 5 * segQ + 3 * segW);

    const int NX = BB * TT * DD;   // 4,194,304
    const int NW = DD * DD;        // 262,144
    cvt_bf16_kernel<<<NX / (256 * 8), 256, 0, stream>>>(x,  xbf, NX);
    cvt_bf16_kernel<<<NW / (256 * 8), 256, 0, stream>>>(wq, wqb, NW);
    cvt_bf16_kernel<<<NW / (256 * 8), 256, 0, stream>>>(wk, wkb, NW);
    cvt_bf16_kernel<<<NW / (256 * 8), 256, 0, stream>>>(wv, wvb, NW);
    cvt_bf16_kernel<<<NW / (256 * 8), 256, 0, stream>>>(wo, wob, NW);

    qkv_proj_kernel<<<1536, 256, 0, stream>>>(xbf, wqb, wkb, wvb, Qbf, Kbf, Vt);
    attn_kernel<<<BB * HH * (TT / 16), 32, 0, stream>>>(Qbf, Kbf, Vt, span, Abf);
    out_proj_kernel<<<512, 256, 0, stream>>>(Abf, wob, out);
    span_loss_kernel<<<1, 1, 0, stream>>>(span, out + (size_t)BB * TT * DD);
}
